// LstmCellModel_75179107549844
// MI455X (gfx1250) — matrix operand order repaired and checked by reading, not compiled, not run on hardware
//
#include <hip/hip_runtime.h>

#define B_SZ 2048
#define T_SZ 2048
#define D_INP 16
#define H1 100
#define H2 50
#define G1N 400   // 4*H1
#define G2N 200   // 4*H2
#define G2P 208   // padded to 13 tiles of 16
#define N1T 25
#define N2T 13
#define H1P 128
#define H2P 64
#define BT 16     // batch rows per workgroup
#define NTHREADS 256
#define NWAVES 8

typedef __attribute__((ext_vector_type(16))) __bf16 v16bf;
typedef __attribute__((ext_vector_type(8)))  float  v8f;

union AFrag { v16bf v; unsigned int u[8]; };

// ---- LDS layout (halves = bf16 stored as u16) ----
#define W0X_HALVES (N1T * 512)        // 25 frags  (K=32 from x)
#define W0H_HALVES (N1T * 4 * 512)    // 100 frags (K=128 from h0)
#define W1X_HALVES (N2T * 4 * 512)    // 52 frags  (K=128 from h0)
#define W1H_HALVES (N2T * 2 * 512)    // 26 frags  (K=64 from h1)
#define XBUF_HALVES (BT * 32)
#define H0_HALVES  (BT * H1P)
#define H1_HALVES  (BT * H2P)
#define HALVES_TOTAL (W0X_HALVES + W0H_HALVES + W1X_HALVES + W1H_HALVES + \
                      XBUF_HALVES + H0_HALVES + H1_HALVES)
#define FLOATS_TOTAL (BT*G1N + BT*G2P + BT*H1 + BT*H2 + G1N + G2N)
#define SMEM_BYTES ((size_t)HALVES_TOTAL * 2 + (size_t)FLOATS_TOTAL * 4)

__device__ __forceinline__ unsigned short f2bf(float f) {
  unsigned int b = __float_as_uint(f);
  b += 0x7FFFu + ((b >> 16) & 1u);          // round-to-nearest-even
  return (unsigned short)(b >> 16);
}
__device__ __forceinline__ float bf2f(unsigned short h) {
  return __uint_as_float(((unsigned int)h) << 16);
}
__device__ __forceinline__ float sigf(float x) { return 1.f / (1.f + __expf(-x)); }
__device__ __forceinline__ float tanh_fast(float x) {
  float e = __expf(-2.f * fabsf(x));
  float t = (1.f - e) / (1.f + e);
  return copysignf(t, x);
}

// A-fragment (16x32 bf16) load from row-major LDS buffer [16 rows, stride halves].
// ISA layout: lanes 0-15: M=lane, v0..3 -> K=0..7 (pairs), v4..7 -> K=16..23;
//             lanes 16-31: M=lane-16, v0..3 -> K=8..15, v4..7 -> K=24..31.
__device__ __forceinline__ v16bf load_a(const unsigned short* base, int lane,
                                        int stride, int kbase) {
  const int m  = lane & 15;
  const int kh = (lane >> 4) << 3;          // 0 or 8
  const unsigned short* row = base + m * stride + kbase;
  AFrag a;
#pragma unroll
  for (int v = 0; v < 4; ++v) {
    a.u[v]     = *(const unsigned int*)(row + kh + 2 * v);
    a.u[v + 4] = *(const unsigned int*)(row + 16 + kh + 2 * v);
  }
  return a.v;
}

// Pre-packed B-fragment: 512 halves per frag; lane holds 16 contiguous halves.
__device__ __forceinline__ v16bf load_b(const unsigned short* base, int frag, int lane) {
  return *(const v16bf*)(base + frag * 512 + lane * 16);
}

// C-tile (16x16 f32) store: VGPR r -> row = r + 8*(lane>=16), col = lane&15.
__device__ __forceinline__ void store_c(float* g, int ldg, int ntile, int lane, v8f acc) {
  const int col = lane & 15, rb = (lane >> 4) << 3;
  float* p = g + rb * ldg + ntile * 16 + col;
#pragma unroll
  for (int r = 0; r < 8; ++r) p[r * ldg] = acc[r];
}

// One layer-0 gate tile: acc = x@W_ih0^T + h0@W_hh0^T for 16 gate columns.
__device__ __forceinline__ v8f l0_tile(const unsigned short* w0x,
                                       const unsigned short* w0h,
                                       int nt, int lane, v16bf ax, const v16bf* ah) {
  v8f acc = {};
  acc = __builtin_amdgcn_wmma_f32_16x16x32_bf16(
          false, ax, false, load_b(w0x, nt, lane), (short)0, acc, false, false);
#pragma unroll
  for (int ks = 0; ks < 4; ++ks)
    acc = __builtin_amdgcn_wmma_f32_16x16x32_bf16(
            false, ah[ks], false, load_b(w0h, nt * 4 + ks, lane),
            (short)0, acc, false, false);
  return acc;
}

// One layer-1 gate tile: acc = h0@W_ih1^T + h1@W_hh1^T for 16 gate columns.
__device__ __forceinline__ v8f l1_tile(const unsigned short* w1x,
                                       const unsigned short* w1h,
                                       int nt, int lane,
                                       const v16bf* ah0, const v16bf* ah1) {
  v8f acc = {};
#pragma unroll
  for (int ks = 0; ks < 4; ++ks)
    acc = __builtin_amdgcn_wmma_f32_16x16x32_bf16(
            false, ah0[ks], false, load_b(w1x, nt * 4 + ks, lane),
            (short)0, acc, false, false);
#pragma unroll
  for (int ks = 0; ks < 2; ++ks)
    acc = __builtin_amdgcn_wmma_f32_16x16x32_bf16(
            false, ah1[ks], false, load_b(w1h, nt * 2 + ks, lane),
            (short)0, acc, false, false);
  return acc;
}

__global__ void __launch_bounds__(NTHREADS)
lstm2_wmma_kernel(const float* __restrict__ x,
                  const float* __restrict__ W_ih0, const float* __restrict__ W_hh0,
                  const float* __restrict__ b_ih0, const float* __restrict__ b_hh0,
                  const float* __restrict__ W_ih1, const float* __restrict__ W_hh1,
                  const float* __restrict__ b_ih1, const float* __restrict__ b_hh1,
                  const float* __restrict__ W_fc,  const float* __restrict__ b_fc,
                  float* __restrict__ out)
{
  extern __shared__ unsigned char smem_raw[];
  unsigned short* w0x = (unsigned short*)smem_raw;
  unsigned short* w0h = w0x + W0X_HALVES;
  unsigned short* w1x = w0h + W0H_HALVES;
  unsigned short* w1h = w1x + W1X_HALVES;
  unsigned short* xb  = w1h + W1H_HALVES;
  unsigned short* h0b = xb  + XBUF_HALVES;
  unsigned short* h1b = h0b + H0_HALVES;
  float* g0    = (float*)(h1b + H1_HALVES);
  float* g1    = g0 + BT * G1N;
  float* c0s   = g1 + BT * G2P;
  float* c1s   = c0s + BT * H1;
  float* bias0 = c1s + BT * H2;
  float* bias1 = bias0 + G1N;

  const int tid  = threadIdx.x;
  const int lane = tid & 31;
  // Wave index is uniform within a wave: make it scalar so tile loops use
  // s_cbranch instead of EXEC-masked VALU loops.
  const int wv   = __builtin_amdgcn_readfirstlane(tid >> 5);
  const int row0 = blockIdx.x * BT;

  // ---------- init: pack weights into bf16 WMMA B-fragments ----------
  for (int idx = tid; idx < W0X_HALVES; idx += NTHREADS) {
    int frag = idx >> 9, r = idx & 511, l = r >> 4, j = r & 15;
    int n = frag * 16 + (l & 15);
    int k = ((l >> 4) << 3) + (j & 7) + ((j >> 3) << 4); // 0..31
    w0x[idx] = f2bf(k < D_INP ? W_ih0[n * D_INP + k] : 0.f);
  }
  for (int idx = tid; idx < W0H_HALVES; idx += NTHREADS) {
    int frag = idx >> 9, r = idx & 511, l = r >> 4, j = r & 15;
    int nt = frag >> 2, ks = frag & 3;
    int n = nt * 16 + (l & 15);
    int k = ks * 32 + ((l >> 4) << 3) + (j & 7) + ((j >> 3) << 4);
    w0h[idx] = f2bf(k < H1 ? W_hh0[n * H1 + k] : 0.f);
  }
  for (int idx = tid; idx < W1X_HALVES; idx += NTHREADS) {
    int frag = idx >> 9, r = idx & 511, l = r >> 4, j = r & 15;
    int nt = frag >> 2, ks = frag & 3;
    int n = nt * 16 + (l & 15);
    int k = ks * 32 + ((l >> 4) << 3) + (j & 7) + ((j >> 3) << 4);
    w1x[idx] = f2bf((n < G2N && k < H1) ? W_ih1[n * H1 + k] : 0.f);
  }
  for (int idx = tid; idx < W1H_HALVES; idx += NTHREADS) {
    int frag = idx >> 9, r = idx & 511, l = r >> 4, j = r & 15;
    int nt = frag >> 1, ks = frag & 1;
    int n = nt * 16 + (l & 15);
    int k = ks * 32 + ((l >> 4) << 3) + (j & 7) + ((j >> 3) << 4);
    w1h[idx] = f2bf((n < G2N && k < H2) ? W_hh1[n * H2 + k] : 0.f);
  }
  for (int idx = tid; idx < G1N; idx += NTHREADS) bias0[idx] = b_ih0[idx] + b_hh0[idx];
  for (int idx = tid; idx < G2N; idx += NTHREADS) bias1[idx] = b_ih1[idx] + b_hh1[idx];
  for (int idx = tid; idx < H0_HALVES;  idx += NTHREADS) h0b[idx] = 0;
  for (int idx = tid; idx < H1_HALVES;  idx += NTHREADS) h1b[idx] = 0;
  for (int idx = tid; idx < XBUF_HALVES; idx += NTHREADS) xb[idx] = 0;
  for (int idx = tid; idx < BT * H1; idx += NTHREADS) c0s[idx] = 0.f;
  for (int idx = tid; idx < BT * H2; idx += NTHREADS) c1s[idx] = 0.f;
  __syncthreads();

  // ---------- time scan ----------
  for (int t = 0; t < T_SZ; ++t) {
    // stage x_t: 64 threads x float4 = 16 rows x 16 features
    if (tid < 64) {
      int r = tid >> 2, d4 = tid & 3;
      size_t gi = ((size_t)(row0 + r) * T_SZ + t) * D_INP + d4 * 4;
      float4 v = *(const float4*)(x + gi);
      uint2 pk;
      pk.x = (unsigned int)f2bf(v.x) | ((unsigned int)f2bf(v.y) << 16);
      pk.y = (unsigned int)f2bf(v.z) | ((unsigned int)f2bf(v.w) << 16);
      *(uint2*)(xb + r * 32 + d4 * 4) = pk;
      if (t + 1 < T_SZ) __builtin_prefetch(x + gi + D_INP, 0, 0);
    }
    __syncthreads();

    // ---- layer 0: gates0[16,400] = x_t @ W_ih0^T + h0 @ W_hh0^T ----
    {
      v16bf ax = load_a(xb, lane, 32, 0);
      v16bf ah[4];
#pragma unroll
      for (int ks = 0; ks < 4; ++ks) ah[ks] = load_a(h0b, lane, H1P, ks * 32);
      int nt = wv;
      for (; nt + NWAVES < N1T; nt += 2 * NWAVES) {   // two independent acc chains
        v8f a0 = l0_tile(w0x, w0h, nt, lane, ax, ah);
        v8f a1 = l0_tile(w0x, w0h, nt + NWAVES, lane, ax, ah);
        store_c(g0, G1N, nt, lane, a0);
        store_c(g0, G1N, nt + NWAVES, lane, a1);
      }
      if (nt < N1T) {
        v8f a0 = l0_tile(w0x, w0h, nt, lane, ax, ah);
        store_c(g0, G1N, nt, lane, a0);
      }
    }
    __syncthreads();

    // ---- layer 0 cell update ----
    for (int idx = tid; idx < BT * H1; idx += NTHREADS) {
      int r = idx / H1, u = idx - r * H1;
      const float* gr = g0 + r * G1N;
      float gi = gr[u]          + bias0[u];
      float gf = gr[H1 + u]     + bias0[H1 + u];
      float gg = gr[2 * H1 + u] + bias0[2 * H1 + u];
      float go = gr[3 * H1 + u] + bias0[3 * H1 + u];
      float c  = sigf(gf) * c0s[idx] + sigf(gi) * tanh_fast(gg);
      c0s[idx] = c;
      h0b[r * H1P + u] = f2bf(sigf(go) * tanh_fast(c));
    }
    __syncthreads();

    // ---- layer 1: gates1[16,208] = h0 @ W_ih1^T + h1 @ W_hh1^T ----
    {
      v16bf ah0[4];
#pragma unroll
      for (int ks = 0; ks < 4; ++ks) ah0[ks] = load_a(h0b, lane, H1P, ks * 32);
      v16bf ah1[2];
#pragma unroll
      for (int ks = 0; ks < 2; ++ks) ah1[ks] = load_a(h1b, lane, H2P, ks * 32);
      int nt = wv;
      for (; nt + NWAVES < N2T; nt += 2 * NWAVES) {
        v8f a0 = l1_tile(w1x, w1h, nt, lane, ah0, ah1);
        v8f a1 = l1_tile(w1x, w1h, nt + NWAVES, lane, ah0, ah1);
        store_c(g1, G2P, nt, lane, a0);
        store_c(g1, G2P, nt + NWAVES, lane, a1);
      }
      if (nt < N2T) {
        v8f a0 = l1_tile(w1x, w1h, nt, lane, ah0, ah1);
        store_c(g1, G2P, nt, lane, a0);
      }
    }
    __syncthreads();

    // ---- layer 1 cell update ----
    for (int idx = tid; idx < BT * H2; idx += NTHREADS) {
      int r = idx / H2, u = idx - r * H2;
      const float* gr = g1 + r * G2P;
      float gi = gr[u]          + bias1[u];
      float gf = gr[H2 + u]     + bias1[H2 + u];
      float gg = gr[2 * H2 + u] + bias1[2 * H2 + u];
      float go = gr[3 * H2 + u] + bias1[3 * H2 + u];
      float c  = sigf(gf) * c1s[idx] + sigf(gi) * tanh_fast(gg);
      c1s[idx] = c;
      h1b[r * H2P + u] = f2bf(sigf(go) * tanh_fast(c));
    }
    __syncthreads();
  }

  // ---------- final FC: [16,50] @ [50,1] ----------
  if (tid < BT) {
    float s = b_fc[0];
#pragma unroll 10
    for (int u = 0; u < H2; ++u) s += bf2f(h1b[tid * H2P + u]) * W_fc[u];
    out[row0 + tid] = s;
  }
}

extern "C" void kernel_launch(void* const* d_in, const int* in_sizes, int n_in,
                              void* d_out, int out_size, void* d_ws, size_t ws_size,
                              hipStream_t stream) {
  (void)in_sizes; (void)n_in; (void)out_size; (void)d_ws; (void)ws_size;
  const float* x     = (const float*)d_in[0];
  const float* W_ih0 = (const float*)d_in[1];
  const float* W_hh0 = (const float*)d_in[2];
  const float* b_ih0 = (const float*)d_in[3];
  const float* b_hh0 = (const float*)d_in[4];
  const float* W_ih1 = (const float*)d_in[5];
  const float* W_hh1 = (const float*)d_in[6];
  const float* b_ih1 = (const float*)d_in[7];
  const float* b_hh1 = (const float*)d_in[8];
  const float* W_fc  = (const float*)d_in[9];
  const float* b_fc  = (const float*)d_in[10];
  float* out = (float*)d_out;

  (void)hipFuncSetAttribute((const void*)lstm2_wmma_kernel,
                            hipFuncAttributeMaxDynamicSharedMemorySize,
                            (int)SMEM_BYTES);
  lstm2_wmma_kernel<<<B_SZ / BT, NTHREADS, SMEM_BYTES, stream>>>(
      x, W_ih0, W_hh0, b_ih0, b_hh0, W_ih1, W_hh1, b_ih1, b_hh1, W_fc, b_fc, out);
}
